// Fp8Linear_82145544503904
// MI455X (gfx1250) — 8-bit matrix operand order repaired and checked by reading, compiled, not run on hardware
//
#include <hip/hip_runtime.h>
#include <hip/hip_bf16.h>
#include <stdint.h>

typedef __attribute__((ext_vector_type(16))) int   v16i;
typedef __attribute__((ext_vector_type(8)))  float v8f;

#define QGROUP   128
#define FP8_MAX  448.0f
#define QEPS     1e-4f

#define ROWB     144
#define A_ROWS   256
#define B_ROWS   128
#define ASZ      (A_ROWS * ROWB)
#define BSZ      (B_ROWS * ROWB)
#define SMEM_BYTES (2 * ASZ + 2 * BSZ)

__device__ __forceinline__ float bf16_lo(uint32_t w) { return __uint_as_float(w << 16); }
__device__ __forceinline__ float bf16_hi(uint32_t w) { return __uint_as_float(w & 0xFFFF0000u); }

#if __has_builtin(__builtin_amdgcn_cvt_pk_fp8_f32)
__device__ __forceinline__ uint32_t pk4_fp8(float f0, float f1, float f2, float f3) {
  int r = 0;
  r = __builtin_amdgcn_cvt_pk_fp8_f32(f0, f1, r, false);
  r = __builtin_amdgcn_cvt_pk_fp8_f32(f2, f3, r, true);
  return (uint32_t)r;
}
#else
__device__ __forceinline__ uint8_t sw_e4m3(float f) {
  float a = fabsf(f);
  uint8_t sign = (f < 0.0f) ? 0x80 : 0x00;
  if (a >= FP8_MAX) return sign | 0x7E;
  if (a < 0.015625f) { int q = (int)rintf(a * 512.0f); return sign | (uint8_t)q; }
  int e; float m = frexpf(a, &e);
  int E = e - 1 + 7;
  int q = (int)rintf(m * 16.0f);
  if (q == 16) { q = 8; ++E; }
  if (E > 15) return sign | 0x7E;
  return sign | (uint8_t)((E << 3) | (q - 8));
}
__device__ __forceinline__ uint32_t pk4_fp8(float f0, float f1, float f2, float f3) {
  return (uint32_t)sw_e4m3(f0) | ((uint32_t)sw_e4m3(f1) << 8) |
         ((uint32_t)sw_e4m3(f2) << 16) | ((uint32_t)sw_e4m3(f3) << 24);
}
#endif

#define LDS_OFF(p) ((uint32_t)(size_t)(__attribute__((address_space(3))) const void*)(p))

__device__ __forceinline__ void async_b128(const uint8_t* g, const uint8_t* l) {
  asm volatile("global_load_async_to_lds_b128 %0, %1, off"
               :: "v"(LDS_OFF(l)), "v"((unsigned long long)(size_t)g)
               : "memory");
}

__global__ __launch_bounds__(256) void quant_act_fp8(
    const __hip_bfloat16* __restrict__ X, uint8_t* __restrict__ X8,
    float* __restrict__ Ascale, int M, int K) {
  const int KG   = K / QGROUP;
  const int gid  = (int)((blockIdx.x * blockDim.x + threadIdx.x) >> 5);
  const int lane = threadIdx.x & 31;
  const int m    = gid / KG;
  const int kg   = gid % KG;
  if (m >= M) return;

  const uint32_t* xp =
      (const uint32_t*)(X + (size_t)m * K + (size_t)kg * QGROUP) + lane * 2;
  const uint32_t w0 = xp[0], w1 = xp[1];
  float f0 = bf16_lo(w0), f1 = bf16_hi(w0), f2 = bf16_lo(w1), f3 = bf16_hi(w1);

  float amax = fmaxf(fmaxf(fabsf(f0), fabsf(f1)), fmaxf(fabsf(f2), fabsf(f3)));
#pragma unroll
  for (int off = 16; off > 0; off >>= 1)
    amax = fmaxf(amax, __shfl_xor(amax, off, 32));

  const float scale = fmaxf(amax, QEPS) * (1.0f / FP8_MAX);
  const float inv   = 1.0f / scale;

  f0 = fminf(fmaxf(f0 * inv, -FP8_MAX), FP8_MAX);
  f1 = fminf(fmaxf(f1 * inv, -FP8_MAX), FP8_MAX);
  f2 = fminf(fmaxf(f2 * inv, -FP8_MAX), FP8_MAX);
  f3 = fminf(fmaxf(f3 * inv, -FP8_MAX), FP8_MAX);

  ((uint32_t*)(X8 + (size_t)m * K + (size_t)kg * QGROUP))[lane] = pk4_fp8(f0, f1, f2, f3);
  if (lane == 0) Ascale[(size_t)kg * M + m] = scale;
}

__global__ __launch_bounds__(256, 1) void fp8_gemm_bias_bf16(
    const uint8_t* __restrict__ A8,
    const float*   __restrict__ Ascale,
    const uint8_t* __restrict__ B8,
    const float*   __restrict__ Bscale,
    const float*   __restrict__ bias,
    __hip_bfloat16* __restrict__ Out,
    int M, int N, int K) {
  extern __shared__ uint8_t smem[];

  const int KG   = K / QGROUP;
  const int tid  = threadIdx.x;
  const int lane = tid & 31;
  const int wave = tid >> 5;
  const int wm   = wave >> 1;
  const int wn   = wave & 1;
  const int rowA0 = blockIdx.y * 256;
  const int colB0 = blockIdx.x * 128;
  const int hi   = lane >> 4;
  const int lm   = lane & 15;

  v8f acc[4][4];
#pragma unroll
  for (int nt = 0; nt < 4; ++nt) {
    const float bv = bias[colB0 + wn * 64 + nt * 16 + lm];
#pragma unroll
    for (int mt = 0; mt < 4; ++mt)
#pragma unroll
      for (int i = 0; i < 8; ++i) acc[mt][nt][i] = bv;
  }

  auto copy_tiles = [&](int kg, int buf) {
    const int kbase = kg * QGROUP;
    uint8_t* sa = smem + (size_t)buf * ASZ;
    uint8_t* sb = smem + 2 * ASZ + (size_t)buf * BSZ;
#pragma unroll
    for (int i = 0; i < 8; ++i) {
      const int c = tid + 256 * i;
      const int r = c >> 3, ch = c & 7;
      async_b128(A8 + (size_t)(rowA0 + r) * K + kbase + ch * 16,
                 sa + r * ROWB + ch * 16);
    }
#pragma unroll
    for (int i = 0; i < 4; ++i) {
      const int c = tid + 256 * i;
      const int r = c >> 3, ch = c & 7;
      async_b128(B8 + (size_t)(colB0 + r) * K + kbase + ch * 16,
                 sb + r * ROWB + ch * 16);
    }
  };

  copy_tiles(0, 0);

  for (int kg = 0; kg < KG; ++kg) {
    const int cur = kg & 1;
    const bool more = (kg + 1 < KG);
    if (more) copy_tiles(kg + 1, cur ^ 1);

    if (more) asm volatile("s_wait_asynccnt 0xC" ::: "memory");
    else      asm volatile("s_wait_asynccnt 0x0" ::: "memory");
    __syncthreads();

    uint8_t* sac = smem + (size_t)cur * ASZ;
    uint8_t* sbc = smem + 2 * ASZ + (size_t)cur * BSZ;

    v16i bfrag[4];
#pragma unroll
    for (int nt = 0; nt < 4; ++nt) {
      const uint8_t* bp = sbc + (wn * 64 + nt * 16 + lm) * ROWB + 16 * hi;
      union { v16i v; uint4 q[4]; } ub;
#pragma unroll
      for (int g = 0; g < 4; ++g) ub.q[g] = *(const uint4*)(bp + 32 * g);
      bfrag[nt] = ub.v;
    }

    const float sBs = Bscale[(size_t)blockIdx.x * KG + kg];

#pragma unroll
    for (int mt = 0; mt < 4; ++mt) {
      const uint8_t* ap = sac + (wm * 64 + mt * 16 + lm) * ROWB + 16 * hi;
      union { v16i v; uint2 d[8]; } ua;
#pragma unroll
      for (int p = 0; p < 8; ++p) ua.d[p] = *(const uint2*)(ap + 32 * (p >> 1) + 8 * (p & 1));

      const float* sap = Ascale + (size_t)kg * M + rowA0 + wm * 64 + mt * 16 + 8 * hi;
      union { v8f v; float4 f[2]; } us;
      us.f[0] = *(const float4*)(sap);
      us.f[1] = *(const float4*)(sap + 4);
      const v8f sv = us.v * sBs;

      v8f d[4];
#pragma unroll
      for (int nt = 0; nt < 4; ++nt) {
        const v8f z = (v8f)0.0f;
        d[nt] = __builtin_amdgcn_wmma_f32_16x16x128_fp8_fp8(
            ua.v, bfrag[nt], (short)0, z, false, false);
      }
#pragma unroll
      for (int nt = 0; nt < 4; ++nt) acc[mt][nt] += d[nt] * sv;
    }

    __syncthreads();
  }

#pragma unroll
  for (int nt = 0; nt < 4; ++nt) {
    const int n = colB0 + wn * 64 + nt * 16 + lm;
#pragma unroll
    for (int mt = 0; mt < 4; ++mt) {
      const int rbase = rowA0 + wm * 64 + mt * 16 + 8 * hi;
#pragma unroll
      for (int v = 0; v < 8; ++v) {
        Out[(size_t)(rbase + v) * N + n] = __float2bfloat16(acc[mt][nt][v]);
      }
    }
  }
}

extern "C" void kernel_launch(void* const* d_in, const int* in_sizes, int n_in,
                              void* d_out, int out_size, void* d_ws, size_t ws_size,
                              hipStream_t stream) {
  const __hip_bfloat16* inp      = (const __hip_bfloat16*)d_in[0];
  const uint8_t*        weight   = (const uint8_t*)d_in[1];
  const float*          w_scales = (const float*)d_in[2];
  const float*          bias     = (const float*)d_in[3];
  __hip_bfloat16*       out      = (__hip_bfloat16*)d_out;

  const int N = in_sizes[3];
  const int K = in_sizes[1] / N;
  const int M = in_sizes[0] / K;
  const int KG = K / QGROUP;

  uint8_t* x8     = (uint8_t*)d_ws;
  float*   ascale = (float*)((uint8_t*)d_ws + (size_t)M * K);

  {
    const long long total_threads = (long long)M * KG * 32;
    const int block = 256;
    const int grid  = (int)((total_threads + block - 1) / block);
    quant_act_fp8<<<grid, block, 0, stream>>>(inp, x8, ascale, M, K);
  }
  {
    dim3 grid(N / 128, M / 256);
    fp8_gemm_bias_bf16<<<grid, 256, SMEM_BYTES, stream>>>(
        x8, ascale, weight, w_scales, bias, out, M, N, K);
  }
}
